// SelfAttention_55954833932866
// MI455X (gfx1250) — matrix operand order repaired and checked by reading, not compiled, not run on hardware
//
#include <hip/hip_runtime.h>
#include <hip/hip_bf16.h>

typedef _Float16 half_t;
typedef __attribute__((ext_vector_type(16))) _Float16 v16h;
typedef __attribute__((ext_vector_type(8)))  _Float16 v8h;
typedef __attribute__((ext_vector_type(8)))  float    v8f;

#define Bq 2
#define Tq 2048
#define Dq 1024
#define Hq 16
#define HDq 64
#define BT (Bq * Tq)                 // 4096 rows
#define N_QKV (3 * Dq)               // 3072
#define BHTD (Bq * Hq * Tq * HDq)    // 4194304 elements per tensor

__device__ __forceinline__ v8f wmma16(v16h a, v16h b, v8f c) {
  // V_WMMA_F32_16X16X32_F16 : f16 operands, fp32 accumulate
  return __builtin_amdgcn_wmma_f32_16x16x32_f16(
      /*neg_a=*/false, a, /*neg_b=*/false, b,
      /*c_mod=*/(short)0, c, /*reuse_a=*/false, /*reuse_b=*/false);
}

__device__ __forceinline__ v8h ld8(const half_t* p) { return *(const v8h*)p; }

__device__ __forceinline__ v16h cat8(v8h a, v8h b) {
  v16h r;
#pragma unroll
  for (int i = 0; i < 8; ++i) { r[i] = a[i]; r[8 + i] = b[i]; }
  return r;
}

// A-matrix fragment (16x32 f16, row-major source):
// lane holds row (lane&15); halves 0..7 = K k0+(lane>>4)*8.., halves 8..15 = +16.
__device__ __forceinline__ v16h load_a(const half_t* row_ptr, int k0, int hi) {
  return cat8(ld8(row_ptr + k0 + hi * 8), ld8(row_ptr + k0 + 16 + hi * 8));
}
// B-matrix fragment (32x16 f16, column-contiguous source):
// lane holds column; halves 0..7 = K k0+(lane>>4)*8.., halves 8..15 = +16.
__device__ __forceinline__ v16h load_b(const half_t* col_ptr, int k0, int hi) {
  return cat8(ld8(col_ptr + k0 + hi * 8), ld8(col_ptr + k0 + 16 + hi * 8));
}

// ---------------------------------------------------------------------------
// fp32 -> f16 elementwise convert
// ---------------------------------------------------------------------------
__global__ __launch_bounds__(256) void cvt_f16(const float* __restrict__ in,
                                               half_t* __restrict__ out) {
  const int i = blockIdx.x * 256 + threadIdx.x;
  out[i] = (half_t)in[i];
}

// ---------------------------------------------------------------------------
// [K][N] fp32 -> [N][K] f16 tiled transpose (column-contiguous weights)
// ---------------------------------------------------------------------------
__global__ __launch_bounds__(256) void transpose_cvt(const float* __restrict__ in,
                                                     half_t* __restrict__ out,
                                                     int K, int N) {
  __shared__ float tile[32][33];
  const int tx = threadIdx.x & 31;
  const int ty = threadIdx.x >> 5;  // 0..7
  const int k0 = blockIdx.y * 32;
  const int n0 = blockIdx.x * 32;
#pragma unroll
  for (int j = 0; j < 4; ++j)
    tile[ty + j * 8][tx] = in[(size_t)(k0 + ty + j * 8) * N + n0 + tx];
  __syncthreads();
#pragma unroll
  for (int j = 0; j < 4; ++j)
    out[(size_t)(n0 + ty + j * 8) * K + k0 + tx] = (half_t)tile[tx][ty + j * 8];
}

// ---------------------------------------------------------------------------
// Pipelined 32x64 wave-tile GEMM core: ping-pong double-buffered fragments,
// 2x-unrolled k loop so buffers alternate roles with no register copies.
// acc[2][4] += A(32xK) @ B(Kx64)
// ---------------------------------------------------------------------------
__device__ __forceinline__ void gemm_core(const half_t* arow0, const half_t* arow1,
                                          const half_t* const bcol[4], int hi,
                                          v8f acc[2][4]) {
  v16h aA0 = load_a(arow0, 0, hi);
  v16h aA1 = load_a(arow1, 0, hi);
  v16h bA[4];
#pragma unroll
  for (int n = 0; n < 4; ++n) bA[n] = load_b(bcol[n], 0, hi);

#pragma unroll 1
  for (int k0 = 0; k0 < Dq; k0 += 64) {
    // prefetch odd stage (k0+32) into B buffers
    v16h aB0 = load_a(arow0, k0 + 32, hi);
    v16h aB1 = load_a(arow1, k0 + 32, hi);
    v16h bB[4];
#pragma unroll
    for (int n = 0; n < 4; ++n) bB[n] = load_b(bcol[n], k0 + 32, hi);
    // compute even stage from A buffers
#pragma unroll
    for (int n = 0; n < 4; ++n) {
      acc[0][n] = wmma16(aA0, bA[n], acc[0][n]);
      acc[1][n] = wmma16(aA1, bA[n], acc[1][n]);
    }
    // prefetch next even stage (k0+64) into A buffers
    if (k0 + 64 < Dq) {
      aA0 = load_a(arow0, k0 + 64, hi);
      aA1 = load_a(arow1, k0 + 64, hi);
#pragma unroll
      for (int n = 0; n < 4; ++n) bA[n] = load_b(bcol[n], k0 + 64, hi);
    }
    // compute odd stage from B buffers
#pragma unroll
    for (int n = 0; n < 4; ++n) {
      acc[0][n] = wmma16(aB0, bB[n], acc[0][n]);
      acc[1][n] = wmma16(aB1, bB[n], acc[1][n]);
    }
  }
}

// ---------------------------------------------------------------------------
// qkv = xh @ WqkvT^T + bqkv -> Q,K [B,H,T,HD] f16 and V [B,H,HD,T] f16.
// ---------------------------------------------------------------------------
__global__ __launch_bounds__(128) void qkv_gemm(const half_t* __restrict__ xh,
                                                const half_t* __restrict__ Wt,
                                                const float* __restrict__ bias,
                                                half_t* __restrict__ qh,
                                                half_t* __restrict__ kh,
                                                half_t* __restrict__ vt) {
  const int lane = threadIdx.x & 31;
  const int wave = threadIdx.x >> 5;
  const int lr = lane & 15;
  const int hi = lane >> 4;
  const int m0 = blockIdx.x * 32;
  const int n_base = (blockIdx.y * 4 + wave) * 64;

  const half_t* arow0 = xh + (size_t)(m0 + lr) * Dq;
  const half_t* arow1 = arow0 + (size_t)16 * Dq;
  const half_t* bcol[4];
#pragma unroll
  for (int n = 0; n < 4; ++n) bcol[n] = Wt + (size_t)(n_base + n * 16 + lr) * Dq;

  v8f acc[2][4] = {};
  gemm_core(arow0, arow1, bcol, hi, acc);

#pragma unroll
  for (int n = 0; n < 4; ++n) {
    const int cg = n_base + n * 16 + lr;  // 0..3071
    const float bv = bias[cg];
    const int which = cg >> 10;           // 0=q 1=k 2=v
    const int rest = cg & 1023;
    const int h = rest >> 6;
    const int hd = rest & 63;
#pragma unroll
    for (int a = 0; a < 2; ++a) {
#pragma unroll
      for (int v = 0; v < 8; ++v) {
        const int r = m0 + a * 16 + v + hi * 8;  // global row in [0,4096)
        const int b_ = r >> 11;
        const int t = r & (Tq - 1);
        const half_t val = (half_t)(acc[a][n][v] + bv);
        if (which == 0)
          qh[(((size_t)(b_ * Hq + h)) * Tq + t) * HDq + hd] = val;
        else if (which == 1)
          kh[(((size_t)(b_ * Hq + h)) * Tq + t) * HDq + hd] = val;
        else
          vt[(((size_t)(b_ * Hq + h)) * HDq + hd) * Tq + t] = val;  // transposed
      }
    }
  }
}

// ---------------------------------------------------------------------------
// Causal flash attention, one wave per 32-query block per (b,h).
// 32 keys/iter: 8 WMMA for S (two q row-tiles share each K fragment),
// fp32 online softmax via shfl_xor butterflies, P staged through padded LDS,
// 8 WMMA for O += P @ V (both row tiles share each V fragment).
// ---------------------------------------------------------------------------
__global__ __launch_bounds__(128) void flash_attn(const half_t* __restrict__ qh,
                                                  const half_t* __restrict__ kh,
                                                  const half_t* __restrict__ vts,
                                                  half_t* __restrict__ yh) {
  __shared__ __align__(16) half_t ldsP[4][32 * 40];  // row stride 40 halves
  const int lane = threadIdx.x & 31;
  const int wave = threadIdx.x >> 5;
  const int lr = lane & 15;
  const int hi = lane >> 4;
  const int qb = blockIdx.x * 4 + wave;  // 32-query block, 0..63
  const int bh = blockIdx.y;             // 0..31

  const half_t* Q  = qh  + (size_t)bh * Tq * HDq;   // [T][64]
  const half_t* K  = kh  + (size_t)bh * Tq * HDq;   // [T][64]
  const half_t* Vt = vts + (size_t)bh * HDq * Tq;   // [64][T]

  // Register-resident Q fragments: 2 row tiles x 2 K=32 chunks
  v16h qa[2][2];
#pragma unroll
  for (int a = 0; a < 2; ++a) {
    const half_t* qrow = Q + (size_t)(qb * 32 + a * 16 + lr) * HDq;
    qa[a][0] = load_a(qrow, 0, hi);
    qa[a][1] = load_a(qrow, 32, hi);
  }

  float m_i[2][8], l_i[2][8];
  v8f o[2][4] = {};
#pragma unroll
  for (int a = 0; a < 2; ++a)
#pragma unroll
    for (int v = 0; v < 8; ++v) { m_i[a][v] = -1e30f; l_i[a][v] = 0.0f; }

  half_t* lp = ldsP[wave];
  const int nblk = qb + 1;  // 32-key blocks, causal

#pragma unroll 1
  for (int kb = 0; kb < nblk; ++kb) {
    const int key0 = kb * 32;

    // ---- S = Q K^T : four 16x16 score tiles (2 row x 2 key sub-tiles) ----
    v8f sc[2][2] = {};
#pragma unroll
    for (int n2 = 0; n2 < 2; ++n2) {
      const half_t* krow = K + (size_t)(key0 + n2 * 16 + lr) * HDq;
      const v16h b0 = load_b(krow, 0, hi);   // head-dim 0..31
      const v16h b1 = load_b(krow, 32, hi);  // head-dim 32..63
#pragma unroll
      for (int a = 0; a < 2; ++a) {
        sc[a][n2] = wmma16(qa[a][0], b0, sc[a][n2]);
        sc[a][n2] = wmma16(qa[a][1], b1, sc[a][n2]);
      }
    }

    // ---- fp32 online softmax per row tile ----
#pragma unroll
    for (int a = 0; a < 2; ++a) {
      const int qti = 2 * qb + a;
      const int kti0 = 2 * kb, kti1 = 2 * kb + 1;
      float p0a[8], p1a[8], alpha[8];
#pragma unroll
      for (int v = 0; v < 8; ++v) {
        const int row = v + hi * 8;       // local query row in tile
        float s0 = sc[a][0][v] * 0.125f;  // HD^-0.5
        float s1 = sc[a][1][v] * 0.125f;
        if (kti0 == qti) { if (lr > row) s0 = -1e30f; }
        else if (kti0 > qti) s0 = -1e30f;
        if (kti1 == qti) { if (lr > row) s1 = -1e30f; }
        else if (kti1 > qti) s1 = -1e30f;
        float rm = fmaxf(s0, s1);
        rm = fmaxf(rm, __shfl_xor(rm, 1, 32));
        rm = fmaxf(rm, __shfl_xor(rm, 2, 32));
        rm = fmaxf(rm, __shfl_xor(rm, 4, 32));
        rm = fmaxf(rm, __shfl_xor(rm, 8, 32));
        const float mn = fmaxf(m_i[a][v], rm);
        alpha[v] = __expf(m_i[a][v] - mn);
        const float p0 = __expf(s0 - mn);
        const float p1 = __expf(s1 - mn);
        float rs = p0 + p1;
        rs += __shfl_xor(rs, 1, 32);
        rs += __shfl_xor(rs, 2, 32);
        rs += __shfl_xor(rs, 4, 32);
        rs += __shfl_xor(rs, 8, 32);
        l_i[a][v] = l_i[a][v] * alpha[v] + rs;
        m_i[a][v] = mn;
        p0a[v] = p0;
        p1a[v] = p1;
      }
      // rescale running output + stage P (rows a*16.. in padded LDS)
#pragma unroll
      for (int n = 0; n < 4; ++n)
#pragma unroll
        for (int v = 0; v < 8; ++v) o[a][n][v] *= alpha[v];
#pragma unroll
      for (int v = 0; v < 8; ++v) {
        lp[(a * 16 + v + hi * 8) * 40 + lr] = (half_t)p0a[v];
        lp[(a * 16 + v + hi * 8) * 40 + 16 + lr] = (half_t)p1a[v];
      }
    }
    __asm__ volatile("" ::: "memory");  // DS ops in-order per wave; block reordering

    // A fragments of P (16x32 each), conflict-free ds_load_b128
    const half_t* pr0 = lp + lr * 40;
    const half_t* pr1 = lp + (16 + lr) * 40;
    const v16h ap0 = cat8(ld8(pr0 + hi * 8), ld8(pr0 + 16 + hi * 8));
    const v16h ap1 = cat8(ld8(pr1 + hi * 8), ld8(pr1 + 16 + hi * 8));

    // ---- O += P @ V : both row tiles share each V fragment ----
#pragma unroll
    for (int n = 0; n < 4; ++n) {
      const half_t* vcol = Vt + (size_t)(n * 16 + lr) * Tq + key0;
      const v16h bv = cat8(ld8(vcol + hi * 8), ld8(vcol + 16 + hi * 8));
      o[0][n] = wmma16(ap0, bv, o[0][n]);
      o[1][n] = wmma16(ap1, bv, o[1][n]);
    }
    __asm__ volatile("" ::: "memory");
  }

  // ---- normalize, store y as f16 [B,T,D] ----
  const int b_ = bh >> 4;
  const int h = bh & 15;
#pragma unroll
  for (int a = 0; a < 2; ++a) {
#pragma unroll
    for (int v = 0; v < 8; ++v) {
      const float inv = 1.0f / l_i[a][v];
      const int t = qb * 32 + a * 16 + v + hi * 8;
      half_t* yrow = yh + ((size_t)(b_ * Tq + t)) * Dq + h * HDq;
#pragma unroll
      for (int n = 0; n < 4; ++n) yrow[n * 16 + lr] = (half_t)(o[a][n][v] * inv);
    }
  }
}

// ---------------------------------------------------------------------------
// out = yh @ WoutT^T + bout (fp32 output)
// ---------------------------------------------------------------------------
__global__ __launch_bounds__(128) void out_gemm(const half_t* __restrict__ yh,
                                                const half_t* __restrict__ Wt,
                                                const float* __restrict__ bias,
                                                float* __restrict__ out) {
  const int lane = threadIdx.x & 31;
  const int wave = threadIdx.x >> 5;
  const int lr = lane & 15;
  const int hi = lane >> 4;
  const int m0 = blockIdx.x * 32;
  const int n_base = (blockIdx.y * 4 + wave) * 64;

  const half_t* arow0 = yh + (size_t)(m0 + lr) * Dq;
  const half_t* arow1 = arow0 + (size_t)16 * Dq;
  const half_t* bcol[4];
#pragma unroll
  for (int n = 0; n < 4; ++n) bcol[n] = Wt + (size_t)(n_base + n * 16 + lr) * Dq;

  v8f acc[2][4] = {};
  gemm_core(arow0, arow1, bcol, hi, acc);

#pragma unroll
  for (int n = 0; n < 4; ++n) {
    const int col = n_base + n * 16 + lr;
    const float bv = bias[col];
#pragma unroll
    for (int a = 0; a < 2; ++a)
#pragma unroll
      for (int v = 0; v < 8; ++v) {
        const int r = m0 + a * 16 + v + hi * 8;
        out[(size_t)r * Dq + col] = acc[a][n][v] + bv;
      }
  }
}

// ---------------------------------------------------------------------------
extern "C" void kernel_launch(void* const* d_in, const int* in_sizes, int n_in,
                              void* d_out, int out_size, void* d_ws, size_t ws_size,
                              hipStream_t stream) {
  const float* x    = (const float*)d_in[0];
  const float* Wqkv = (const float*)d_in[1];
  const float* bqkv = (const float*)d_in[2];
  const float* Wout = (const float*)d_in[3];
  const float* bout = (const float*)d_in[4];
  float* out = (float*)d_out;

  // workspace carve (f16 buffers), ~48 MB total
  char* w = (char*)d_ws;
  half_t* xh    = (half_t*)w; w += (size_t)BT * Dq * sizeof(half_t);      // 8 MB
  half_t* WqkvT = (half_t*)w; w += (size_t)N_QKV * Dq * sizeof(half_t);   // 6 MB
  half_t* WoutT = (half_t*)w; w += (size_t)Dq * Dq * sizeof(half_t);      // 2 MB
  half_t* qh    = (half_t*)w; w += (size_t)BHTD * sizeof(half_t);         // 8 MB
  half_t* kh    = (half_t*)w; w += (size_t)BHTD * sizeof(half_t);         // 8 MB
  half_t* vt    = (half_t*)w; w += (size_t)BHTD * sizeof(half_t);         // 8 MB
  half_t* yh    = (half_t*)w;                                             // 8 MB

  // 0) precision/layout prep
  cvt_f16<<<(BT * Dq) / 256, 256, 0, stream>>>(x, xh);
  transpose_cvt<<<dim3(N_QKV / 32, Dq / 32), 256, 0, stream>>>(Wqkv, WqkvT, Dq, N_QKV);
  transpose_cvt<<<dim3(Dq / 32, Dq / 32), 256, 0, stream>>>(Wout, WoutT, Dq, Dq);

  // 1) QKV projection: 4096x3072, 32x64 per wave
  qkv_gemm<<<dim3(BT / 32, N_QKV / 256), 128, 0, stream>>>(xh, WqkvT, bqkv, qh, kh, vt);

  // 2) causal flash attention: 32 queries per wave
  flash_attn<<<dim3((Tq / 32) / 4, Bq * Hq), 128, 0, stream>>>(qh, kh, vt, yh);

  // 3) output projection: 4096x1024, 32x64 per wave
  out_gemm<<<dim3(BT / 32, Dq / 256), 128, 0, stream>>>(yh, WoutT, bout, out);
}
